// GraphER_27960237097164
// MI455X (gfx1250) — matrix operand order repaired and checked by reading, not compiled, not run on hardware
//
#include <hip/hip_runtime.h>

#define HDIM   128
#define NNODES 50000
#define NEDGES 600000
#define NCAND  262144
#define NT16   (NNODES / 16)   // 3125 row tiles

#define LDA 136   // padded LDS stride (halves) for 128-wide rows: 272B -> 4-bank/lane
#define LDB 264   // padded LDS stride (halves) for 256-wide rows: 528B -> 4-bank/lane

typedef __bf16 bf16_t;
typedef __attribute__((ext_vector_type(16))) __bf16 v16bf;
typedef __attribute__((ext_vector_type(8)))  __bf16 v8bf;
typedef __attribute__((ext_vector_type(8)))  float  v8f;
typedef __attribute__((ext_vector_type(4)))  float  v4f;

#if __has_builtin(__builtin_amdgcn_global_load_async_to_lds_b128)
#define HAVE_ASYNC_LDS 1
typedef int v4i_t __attribute__((vector_size(16)));
typedef __attribute__((address_space(1))) v4i_t* as1_v4i;
typedef __attribute__((address_space(3))) v4i_t* as3_v4i;
#endif

// float -> bf16 via native hardware convert (RNE on gfx1250)
__device__ __forceinline__ bf16_t f2bf(float f) { return (bf16_t)f; }

__device__ __forceinline__ v8f wmma_bf16(v16bf a, v16bf b, v8f c) {
  return __builtin_amdgcn_wmma_f32_16x16x32_bf16(false, a, false, b, (short)0, c,
                                                 false, false);
}

__device__ __forceinline__ v16bf cat8(v8bf lo, v8bf hi) {
  return __builtin_shufflevector(lo, hi, 0, 1, 2, 3, 4, 5, 6, 7, 8, 9, 10, 11, 12,
                                 13, 14, 15);
}

__device__ __forceinline__ v8bf pack8(const bf16_t* t) {
  v8bf r;
#pragma unroll
  for (int i = 0; i < 8; ++i) r[i] = t[i];
  return r;
}

__device__ __forceinline__ v16bf pack16(const bf16_t* t) {
  v16bf r;
#pragma unroll
  for (int i = 0; i < 16; ++i) r[i] = t[i];
  return r;
}

// A fragment from a 16-row bf16 LDS tile with row stride ld (halves).
// lane -> row m = lane&15; K halves {kc*32+off..+7, kc*32+16+off..+23}, off=(lane>>4)*8
__device__ __forceinline__ v16bf load_a_lds(const bf16_t* base, int ld, int lane,
                                            int kc) {
  int m = lane & 15;
  int off = (lane >> 4) * 8;
  const bf16_t* p = base + m * ld + kc * 32 + off;
  v8bf lo = *(const v8bf*)p;
  v8bf hi = *(const v8bf*)(p + 16);
  return cat8(lo, hi);
}

// B fragment from LDS-staged W^T [N][K] with row stride ld (halves):
// lane -> col N = j*16 + (lane&15), K {kc*32+off..+7, kc*32+16+off..+23}, off=(lane>>4)*8
__device__ __forceinline__ v16bf load_b_lds(const bf16_t* base, int ld, int lane,
                                            int j, int kc) {
  int n = (j << 4) + (lane & 15);
  int kb = kc * 32 + ((lane >> 4) << 3);
  const bf16_t* p = base + n * ld + kb;
  v8bf lo = *(const v8bf*)p;
  v8bf hi = *(const v8bf*)(p + 16);
  return cat8(lo, hi);
}

// ---- gfx1250 async global->LDS staging (16B per lane), guarded fallback ----
__device__ __forceinline__ void stage_b128(const bf16_t* g, bf16_t* l) {
#ifdef HAVE_ASYNC_LDS
  __builtin_amdgcn_global_load_async_to_lds_b128(
      (as1_v4i)(unsigned long long)g,
      (as3_v4i)(unsigned)(unsigned long long)l, 0, 0);
#else
  *(v8bf*)l = *(const v8bf*)g;
#endif
}

__device__ __forceinline__ void stage_wait() {
#ifdef HAVE_ASYNC_LDS
#if __has_builtin(__builtin_amdgcn_s_wait_asynccnt)
  __builtin_amdgcn_s_wait_asynccnt(0);
#else
  asm volatile("s_wait_asynccnt 0x0" ::: "memory");
#endif
#endif
}

// ---------------- weight prep: convert + transpose to bf16 ----------------
__global__ __launch_bounds__(256) void prep_weights(
    const float* __restrict__ gW1, const float* __restrict__ gW2,
    const float* __restrict__ epW1, bf16_t* __restrict__ w1t,
    bf16_t* __restrict__ w2t, bf16_t* __restrict__ epw1t) {
  int tid = blockIdx.x * 256 + threadIdx.x;
  if (tid < 49152) {  // gin_W1: wt[l][n][k] = W[l][k][n]
    int l = tid >> 14, n = (tid >> 7) & 127, k = tid & 127;
    w1t[tid] = f2bf(gW1[(l << 14) + (k << 7) + n]);
  } else if (tid < 98304) {
    int t = tid - 49152;
    int l = t >> 14, n = (t >> 7) & 127, k = t & 127;
    w2t[t] = f2bf(gW2[(l << 14) + (k << 7) + n]);
  } else if (tid < 131072) {  // ep_W1 rows 256..511 -> epw1t[n][k] (128 x 256)
    int t = tid - 98304;
    int n = t >> 8, k = t & 255;
    epw1t[t] = f2bf(epW1[(long)(256 + k) * HDIM + n]);
  }
}

// ---------------- zero + scatter-add (segment_sum) ----------------
__global__ __launch_bounds__(256) void zero_f4(float* __restrict__ p, int n4) {
  int t = blockIdx.x * 256 + threadIdx.x;
  if (t < n4) {
    v4f z = {0.f, 0.f, 0.f, 0.f};
    ((v4f*)p)[t] = z;
  }
}

__global__ __launch_bounds__(256) void scatter_add(const float* __restrict__ x,
                                                   const int* __restrict__ ei,
                                                   float* __restrict__ agg) {
  int t = blockIdx.x * 256 + threadIdx.x;
  int e = t >> 5;
  if (e >= NEDGES) return;
  int f = (t & 31) << 2;
  int s = ei[e];
  int d = ei[NEDGES + e];
  v4f val = *(const v4f*)(x + (long)s * HDIM + f);
  float* dst = agg + (long)d * HDIM + f;
  atomicAdd(dst + 0, val.x);
  atomicAdd(dst + 1, val.y);
  atomicAdd(dst + 2, val.z);
  atomicAdd(dst + 3, val.w);
}

// -------- fused GIN layer: x' = relu((x+agg)W1+b1)W2+b2, 8 waves/block --------
// LDS: W1t (128xLDA) | W2t (128xLDA) | per-wave A (16xLDA) | per-wave T (16xLDA)
__global__ __launch_bounds__(256) void gin_layer(
    const float* __restrict__ xin, const float* __restrict__ agg,
    const bf16_t* __restrict__ w1t, const bf16_t* __restrict__ w2t,
    const float* __restrict__ b1, const float* __restrict__ b2,
    float* __restrict__ xout) {
  extern __shared__ __align__(16) char smem[];
  bf16_t* lW1 = (bf16_t*)smem;
  bf16_t* lW2 = lW1 + 128 * LDA;
  bf16_t* lAall = lW2 + 128 * LDA;
  bf16_t* lTall = lAall + 8 * 16 * LDA;

  int tid = threadIdx.x;
  // stage cols 0..63 of both 128x128 bf16 weight matrices (async -> LDS)
#pragma unroll
  for (int i = 0; i < 4; ++i) {
    int ch = i * 256 + tid;  // 1024 x 16B chunks per matrix
    int row = ch >> 3, c8 = (ch & 7) << 3;
    stage_b128(w1t + row * HDIM + c8, lW1 + row * LDA + c8);
    stage_b128(w2t + row * HDIM + c8, lW2 + row * LDA + c8);
  }
  stage_wait();

  int lane = tid & 31, wave = tid >> 5;
  int tile = blockIdx.x * 8 + wave;
  bool active = tile < NT16;
  long rowbase = (long)(active ? tile : NT16 - 1) * 16;
  bf16_t* mA = lAall + wave * 16 * LDA;
  bf16_t* mT = lTall + wave * 16 * LDA;

  // stage (x + agg) tile as bf16: lane -> row lane>>1, half-row (lane&1)*64
  int r0 = lane >> 1, c0 = (lane & 1) * 64;
  const float* px = xin + (rowbase + r0) * HDIM + c0;
  const float* pa = agg + (rowbase + r0) * HDIM + c0;
#pragma unroll
  for (int c = 0; c < 64; c += 8) {
    v4f x0 = *(const v4f*)(px + c), x1 = *(const v4f*)(px + c + 4);
    v4f g0 = *(const v4f*)(pa + c), g1 = *(const v4f*)(pa + c + 4);
    bf16_t t8[8];
#pragma unroll
    for (int q = 0; q < 4; ++q) {
      t8[q] = f2bf(x0[q] + g0[q]);
      t8[4 + q] = f2bf(x1[q] + g1[q]);
    }
    *(v8bf*)(mA + r0 * LDA + c0 + c) = pack8(t8);
  }
  __syncthreads();

  v16bf a0 = load_a_lds(mA, LDA, lane, 0), a1 = load_a_lds(mA, LDA, lane, 1),
        a2 = load_a_lds(mA, LDA, lane, 2), a3 = load_a_lds(mA, LDA, lane, 3);

  int n_lo = lane & 15;
  int mrow = (lane >> 4) * 8;

  // stage 1: T = relu(A @ W1 + b1) -> LDS (bf16)
#pragma unroll
  for (int j = 0; j < 8; ++j) {
    v8f acc = {};
    acc = wmma_bf16(a0, load_b_lds(lW1, LDA, lane, j, 0), acc);
    acc = wmma_bf16(a1, load_b_lds(lW1, LDA, lane, j, 1), acc);
    acc = wmma_bf16(a2, load_b_lds(lW1, LDA, lane, j, 2), acc);
    acc = wmma_bf16(a3, load_b_lds(lW1, LDA, lane, j, 3), acc);
    float bias = b1[j * 16 + n_lo];
#pragma unroll
    for (int r = 0; r < 8; ++r)
      mT[(mrow + r) * LDA + j * 16 + n_lo] = f2bf(fmaxf(acc[r] + bias, 0.f));
  }
  __syncthreads();

  v16bf t0 = load_a_lds(mT, LDA, lane, 0), t1 = load_a_lds(mT, LDA, lane, 1),
        t2 = load_a_lds(mT, LDA, lane, 2), t3 = load_a_lds(mT, LDA, lane, 3);

  // stage 2: X' = T @ W2 + b2
#pragma unroll
  for (int j = 0; j < 8; ++j) {
    v8f acc = {};
    acc = wmma_bf16(t0, load_b_lds(lW2, LDA, lane, j, 0), acc);
    acc = wmma_bf16(t1, load_b_lds(lW2, LDA, lane, j, 1), acc);
    acc = wmma_bf16(t2, load_b_lds(lW2, LDA, lane, j, 2), acc);
    acc = wmma_bf16(t3, load_b_lds(lW2, LDA, lane, j, 3), acc);
    if (active) {
      float bias = b2[j * 16 + n_lo];
#pragma unroll
      for (int r = 0; r < 8; ++r)
        xout[(rowbase + mrow + r) * HDIM + j * 16 + n_lo] = acc[r] + bias;
    }
  }
}

// ---------------- constant part: t_embed, first_feat, folded const vec ----------
__global__ __launch_bounds__(128) void make_const(
    const float* __restrict__ x, const int* __restrict__ fe,
    const int* __restrict__ t_i, const float* __restrict__ epW1,
    const float* __restrict__ epb1, const float* __restrict__ teW1,
    const float* __restrict__ teb1, const float* __restrict__ teW2,
    const float* __restrict__ teb2, float* __restrict__ cvec) {
  __shared__ float relu_t[128];
  __shared__ float ff[256];
  __shared__ float temb[128];
  int n = threadIdx.x;
  float tf = (float)t_i[0];
  relu_t[n] = fmaxf(tf * teW1[n] + teb1[n], 0.f);
  int u = fe[0], v = fe[1];
  float xu = x[(long)u * HDIM + n], xv = x[(long)v * HDIM + n];
  ff[n] = xu + xv;
  ff[128 + n] = fabsf(xu - xv);
  __syncthreads();
  float te = teb2[n];
  for (int k = 0; k < 128; ++k) te += relu_t[k] * teW2[k * HDIM + n];
  temb[n] = te;
  __syncthreads();
  float s = epb1[n];
  for (int k = 0; k < 256; ++k) s += ff[k] * epW1[(long)k * HDIM + n];
  for (int k = 0; k < 128; ++k) s += temb[k] * epW1[(long)(512 + k) * HDIM + n];
  cvec[n] = s;
}

// ---------------- edge predictor: LDS-staged B, 8 tiles per wave ----------------
__device__ __forceinline__ void load8(const float* __restrict__ xu,
                                      const float* __restrict__ xv, int k,
                                      bf16_t* o) {
  if (k < 128) {  // x_u + x_v
    v4f u0 = *(const v4f*)(xu + k), u1 = *(const v4f*)(xu + k + 4);
    v4f v0 = *(const v4f*)(xv + k), v1 = *(const v4f*)(xv + k + 4);
#pragma unroll
    for (int q = 0; q < 4; ++q) {
      o[q] = f2bf(u0[q] + v0[q]);
      o[4 + q] = f2bf(u1[q] + v1[q]);
    }
  } else {  // |x_u - x_v|
    int kk = k - 128;
    v4f u0 = *(const v4f*)(xu + kk), u1 = *(const v4f*)(xu + kk + 4);
    v4f v0 = *(const v4f*)(xv + kk), v1 = *(const v4f*)(xv + kk + 4);
#pragma unroll
    for (int q = 0; q < 4; ++q) {
      o[q] = f2bf(fabsf(u0[q] - v0[q]));
      o[4 + q] = f2bf(fabsf(u1[q] - v1[q]));
    }
  }
}

__global__ __launch_bounds__(256) void edge_pred(
    const float* __restrict__ x, const int* __restrict__ cand,
    const bf16_t* __restrict__ epw1t, const float* __restrict__ cvec,
    const float* __restrict__ w2, const float* __restrict__ b2,
    float* __restrict__ out) {
  extern __shared__ __align__(16) char smem[];
  bf16_t* lB = (bf16_t*)smem;  // 128 x LDB halves

  int tid = threadIdx.x;
  // stage cols 0..127 of ep_W1t (128 x 256 bf16) into LDS, rows padded to LDB
#pragma unroll
  for (int i = 0; i < 8; ++i) {
    int ch = i * 256 + tid;  // 2048 x 16B chunks
    int row = ch >> 4, c8 = (ch & 15) << 3;
    stage_b128(epw1t + row * 256 + c8, lB + row * LDB + c8);
  }
  stage_wait();
  __syncthreads();

  int lane = tid & 31, wave = tid >> 5;
  long cbase = (long)blockIdx.x * 1024 + (long)wave * 128;
  int m = lane & 15;
  int off = (lane >> 4) * 8;
  float bb = b2[0];

  for (int tt = 0; tt < 8; ++tt) {
    long c = cbase + tt * 16 + m;
    int u = cand[2 * c], v = cand[2 * c + 1];
    const float* xu = x + (long)u * HDIM;
    const float* xv = x + (long)v * HDIM;

    // prefetch next tile's gathered rows while this tile computes
    if (tt < 7) {
      long cn = c + 16;
      int un = cand[2 * cn], vn = cand[2 * cn + 1];
      const float* pu = x + (long)un * HDIM;
      const float* pv = x + (long)vn * HDIM;
      __builtin_prefetch(pu, 0, 3);
      __builtin_prefetch(pu + 64, 0, 3);
      __builtin_prefetch(pv, 0, 3);
      __builtin_prefetch(pv + 64, 0, 3);
    }

    // build 16x256 A tile (8 K-chunks) in registers
    v16bf a[8];
#pragma unroll
    for (int kc = 0; kc < 8; ++kc) {
      bf16_t tmp[16];
      int k0 = kc * 32 + off;
      load8(xu, xv, k0, tmp);
      load8(xu, xv, k0 + 16, tmp + 8);
      a[kc] = pack16(tmp);
    }

    float partial[8] = {0.f, 0.f, 0.f, 0.f, 0.f, 0.f, 0.f, 0.f};
#pragma unroll
    for (int j = 0; j < 8; ++j) {
      v8f acc = {};
#pragma unroll
      for (int kc = 0; kc < 8; ++kc)
        acc = wmma_bf16(a[kc], load_b_lds(lB, LDB, lane, j, kc), acc);
      int n = j * 16 + (lane & 15);
      float cv = cvec[n];
      float wn = w2[n];
#pragma unroll
      for (int r = 0; r < 8; ++r) partial[r] += fmaxf(acc[r] + cv, 0.f) * wn;
    }
    // reduce over the 16 n-lanes of each half-wave
#pragma unroll
    for (int r = 0; r < 8; ++r) {
      float s = partial[r];
      s += __shfl_xor(s, 1, 16);
      s += __shfl_xor(s, 2, 16);
      s += __shfl_xor(s, 4, 16);
      s += __shfl_xor(s, 8, 16);
      partial[r] = s;
    }
    if ((lane & 15) == 0) {
      long base = cbase + tt * 16 + (lane >> 4) * 8;
#pragma unroll
      for (int r = 0; r < 8; ++r) out[base + r] = partial[r] + bb;
    }
  }
}

// ---------------- host launch ----------------
extern "C" void kernel_launch(void* const* d_in, const int* in_sizes, int n_in,
                              void* d_out, int out_size, void* d_ws, size_t ws_size,
                              hipStream_t stream) {
  const float* x_in = (const float*)d_in[0];
  const int* ei = (const int*)d_in[1];
  const int* fe = (const int*)d_in[2];
  const int* cand = (const int*)d_in[3];
  const int* t_i = (const int*)d_in[4];
  const float* gW1 = (const float*)d_in[5];
  const float* gb1 = (const float*)d_in[6];
  const float* gW2 = (const float*)d_in[7];
  const float* gb2 = (const float*)d_in[8];
  const float* epW1 = (const float*)d_in[9];
  const float* epb1 = (const float*)d_in[10];
  const float* epW2 = (const float*)d_in[11];
  const float* epb2 = (const float*)d_in[12];
  const float* teW1 = (const float*)d_in[13];
  const float* teb1 = (const float*)d_in[14];
  const float* teW2 = (const float*)d_in[15];
  const float* teb2 = (const float*)d_in[16];
  float* out = (float*)d_out;

  char* ws = (char*)d_ws;
  const size_t SZ_X = (size_t)NNODES * HDIM * sizeof(float);  // 25.6 MB
  float* xa = (float*)(ws);
  float* xb = (float*)(ws + SZ_X);
  float* agg = (float*)(ws + 2 * SZ_X);
  bf16_t* w1t = (bf16_t*)(ws + 3 * SZ_X);
  bf16_t* w2t = (bf16_t*)(ws + 3 * SZ_X + 98304);
  bf16_t* epw1t = (bf16_t*)(ws + 3 * SZ_X + 2 * 98304);
  float* cvec = (float*)(ws + 3 * SZ_X + 2 * 98304 + 65536);

  const size_t SMEM_GIN = (size_t)(2 * 128 * LDA + 2 * 8 * 16 * LDA) * sizeof(bf16_t);
  const size_t SMEM_EP = (size_t)(128 * LDB) * sizeof(bf16_t);

  prep_weights<<<512, 256, 0, stream>>>(gW1, gW2, epW1, w1t, w2t, epw1t);

  const float* cur = x_in;
  float* nxt = xa;
  for (int l = 0; l < 3; ++l) {
    zero_f4<<<(NNODES * HDIM / 4 + 255) / 256, 256, 0, stream>>>(agg,
                                                                 NNODES * HDIM / 4);
    scatter_add<<<(NEDGES * 32) / 256, 256, 0, stream>>>(cur, ei, agg);
    gin_layer<<<(NT16 + 7) / 8, 256, SMEM_GIN, stream>>>(
        cur, agg, w1t + l * 16384, w2t + l * 16384, gb1 + l * HDIM,
        gb2 + l * HDIM, nxt);
    cur = nxt;
    nxt = (nxt == xa) ? xb : xa;
  }
  // after 3 layers cur == xa
  make_const<<<1, 128, 0, stream>>>(cur, fe, t_i, epW1, epb1, teW1, teb1, teW2,
                                    teb2, cvec);
  edge_pred<<<NCAND / 1024, 256, SMEM_EP, stream>>>(cur, cand, epw1t, cvec, epW2,
                                                    epb2, out);
}
